// BailingMoEAttention_24266565222511
// MI455X (gfx1250) — matrix operand order repaired and checked by reading, not compiled, not run on hardware
//
#include <hip/hip_runtime.h>
#include <stdint.h>

// ---------------- problem constants (match reference) ----------------
#define B_    2
#define S_    2048
#define HID_  2048
#define NH_   16
#define NKV_  4
#define HD_   128
#define QK_SCALE 0.08838834764831845f   // HD^-0.5
#define ROPE_L2  0.2076205059304602f    // log2(10000)/64

typedef __bf16 bf16;
typedef __attribute__((ext_vector_type(16))) __bf16 v16bf;
typedef __attribute__((ext_vector_type(8)))  __bf16 v8bf;
typedef __attribute__((ext_vector_type(4)))  __bf16 v4bf;
typedef __attribute__((ext_vector_type(8)))  float  v8f;

// Assemble a 16-bit-type WMMA A fragment for one lane.
// A layout (contiguous k order, matching B fills): lanes 0-15 hold row M=lane,
// K = {0..15}; lanes 16-31 hold row M=lane-16, K = {16..31}.
__device__ __forceinline__ v16bf wmma_a_frag(const bf16* rowptr, int hi) {
  const bf16* p = rowptr + (hi ? 16 : 0);
  v8bf lo = *(const v8bf*)(p);        // K base+{0..7} or base+{16..23}
  v8bf h2 = *(const v8bf*)(p + 8);    // K base+{8..15} or base+{24..31}
  v16bf r;
#pragma unroll
  for (int i = 0; i < 8; ++i) { r[i] = lo[i]; r[i + 8] = h2[i]; }
  return r;
}

__device__ __forceinline__ v8f wmma_bf16(v16bf a, v16bf b, v8f c) {
  return __builtin_amdgcn_wmma_f32_16x16x32_bf16(
      /*neg_a=*/false, a, /*neg_b=*/false, b,
      /*c_mod=*/(short)0, c, /*reuse_a=*/false, /*reuse_b=*/false);
}

// CDNA5 async global->LDS DMA (ISA 15.18.3 op 98, tracked by ASYNCcnt).
// VDST lane value = LDS byte address; low 32 bits of a flat shared pointer
// are exactly the LDS offset (aperture mapping, ISA 10.2).
__device__ __forceinline__ void async_copy_b128(void* lds_ptr, const void* gptr) {
  uint32_t l = (uint32_t)(uintptr_t)lds_ptr;
  asm volatile("global_load_async_to_lds_b128 %0, %1, off"
               :: "v"(l), "v"(gptr) : "memory");
}

// ---------------- elementwise prep kernels ----------------
__global__ void cvt_f32_to_bf16(const float* __restrict__ in,
                                bf16* __restrict__ out, int n) {
  int tid = blockIdx.x * blockDim.x + threadIdx.x;
  if (tid < n) out[tid] = (bf16)in[tid];
}

// W [rows, cols] fp32 -> Wt [cols, rows] bf16  (so B-fragments are contiguous)
__global__ void transpose_to_bf16(const float* __restrict__ W,
                                  bf16* __restrict__ Wt, int rows, int cols) {
  int tid = blockIdx.x * blockDim.x + threadIdx.x;
  if (tid >= rows * cols) return;
  int c = tid % cols, r = tid / cols;
  Wt[(size_t)c * rows + r] = (bf16)W[tid];
}

// V fp32 [B,S,NKV*HD] -> bf16 [B,NKV,HD,S]  (keys contiguous per dim)
__global__ void v_transpose_bf16(const float* __restrict__ V,
                                 bf16* __restrict__ Vt) {
  int tid = blockIdx.x * blockDim.x + threadIdx.x;
  if (tid >= B_ * S_ * NKV_ * HD_) return;
  int d  = tid % HD_;  int t1 = tid / HD_;
  int h  = t1 % NKV_;  int t2 = t1 / NKV_;
  int s  = t2 % S_;    int b  = t2 / S_;
  Vt[(((size_t)(b * NKV_ + h)) * HD_ + d) * S_ + s] = (bf16)V[tid];
}

// ------------- tiled WMMA GEMM: C[M,N] = A[M,K] * Bt[N,K]^T -------------
// 256 threads (8 waves). Workgroup tile: 64(M) x 128(N). K staged in LDS in
// chunks of 64, double buffered, filled with async global->LDS b128 DMA.
// Wave w: mw = w&3 selects 16 M rows, nw = w>>2 selects 64 N cols (4 tiles).
#define TM 64
#define TN 128
#define KC 64
__global__ void wmma_gemm_tiled(const bf16* __restrict__ A,
                                const bf16* __restrict__ Bt,
                                float* __restrict__ C,
                                int M, int N, int K) {
  __shared__ __align__(16) bf16 Atile[2][TM * KC];   // 8 KB per buffer
  __shared__ __align__(16) bf16 Btile[2][TN * KC];   // 16 KB per buffer
  const int tid  = threadIdx.x;
  const int lane = tid & 31;
  const int wv   = tid >> 5;
  const int hi = lane >> 4, lm = lane & 15;
  const int mw = wv & 3, nw = wv >> 2;
  const int m0 = blockIdx.x * TM;
  const int n0 = blockIdx.y * TN;

  v8f acc[4] = {};

  // Stage one KC-wide slab into buffer b: A 64x64 (2 chunks/thread),
  // B 128x64 (4 chunks/thread); 6 async b128 ops per thread.
  auto stage = [&](int kc, int b) {
#pragma unroll
    for (int u = 0; u < 2; ++u) {
      int q = tid * 2 + u;               // 512 chunks of 8 bf16
      int r = q >> 3, c = (q & 7) * 8;
      async_copy_b128(&Atile[b][r * KC + c], A + (size_t)(m0 + r) * K + kc + c);
    }
#pragma unroll
    for (int u = 0; u < 4; ++u) {
      int q = tid * 4 + u;               // 1024 chunks of 8 bf16
      int r = q >> 3, c = (q & 7) * 8;
      async_copy_b128(&Btile[b][r * KC + c], Bt + (size_t)(n0 + r) * K + kc + c);
    }
  };

  stage(0, 0);
  int buf = 0;
  for (int kc = 0; kc < K; kc += KC) {
    if (kc + KC < K) {
      stage(kc + KC, buf ^ 1);
      asm volatile("s_wait_asynccnt 6" ::: "memory");  // keep next slab in flight
    } else {
      asm volatile("s_wait_asynccnt 0" ::: "memory");
    }
    __syncthreads();                                    // slab visible to all waves
#pragma unroll
    for (int s = 0; s < 2; ++s) {                       // two 32-wide sub-chunks
      v16bf a = wmma_a_frag(&Atile[buf][(mw * 16 + lm) * KC + s * 32], hi);
#pragma unroll
      for (int t = 0; t < 4; ++t) {
        const bf16* bp =
            &Btile[buf][(nw * 64 + t * 16 + lm) * KC + s * 32 + hi * 16];
        v16bf bfrag = *(const v16bf*)bp;
        acc[t] = wmma_bf16(a, bfrag, acc[t]);
      }
    }
    __syncthreads();                                    // done reading buf
    buf ^= 1;
  }

  // C layout: lane holds column lm; VGPR r holds row r + 8*hi.
#pragma unroll
  for (int t = 0; t < 4; ++t) {
    int col = n0 + nw * 64 + t * 16 + lm;
#pragma unroll
    for (int r = 0; r < 8; ++r) {
      int m = m0 + mw * 16 + r + 8 * hi;
      C[(size_t)m * N + col] = acc[t][r];
    }
  }
}

// ---------------- per-head RMSNorm + NeoX RoPE (one wave32 per row) ----------------
// X fp32 [B,S,H*HD] -> Y bf16 [B,H,S,HD], scaled by outmul (QK_SCALE for q).
__global__ void norm_rope_kernel(const float* __restrict__ X,
                                 const int* __restrict__ positions,
                                 const float* __restrict__ scale,
                                 bf16* __restrict__ Y,
                                 int H, int nrows, float outmul) {
  int wid  = (blockIdx.x * blockDim.x + threadIdx.x) >> 5;
  int lane = threadIdx.x & 31;
  if (wid >= nrows) return;                  // wave-uniform
  int h  = wid % H;
  int bs = wid / H;
  int s  = bs % S_, b = bs / S_;

  const float* xp = X + (size_t)wid * HD_ + lane * 4;
  float4 v = *(const float4*)xp;
  float ss = v.x * v.x + v.y * v.y + v.z * v.z + v.w * v.w;
#pragma unroll
  for (int m = 16; m >= 1; m >>= 1) ss += __shfl_xor(ss, m, 32);
  float rn = rsqrtf(ss * (1.0f / (float)HD_) + 1e-6f);

  float xs[4];
  xs[0] = v.x * rn * scale[lane * 4 + 0];
  xs[1] = v.y * rn * scale[lane * 4 + 1];
  xs[2] = v.z * rn * scale[lane * 4 + 2];
  xs[3] = v.w * rn * scale[lane * 4 + 3];

  float ot[4];
#pragma unroll
  for (int j = 0; j < 4; ++j) ot[j] = __shfl_xor(xs[j], 16, 32);

  float pos = (float)positions[b * S_ + s];
  v4bf out;
#pragma unroll
  for (int j = 0; j < 4; ++j) {
    int i  = lane * 4 + j;
    int jf = (i < 64) ? i : i - 64;
    float invf = exp2f(-(float)jf * ROPE_L2);
    float ang  = pos * invf;
    float c = cosf(ang), sn = sinf(ang);
    float o = (i < 64) ? (xs[j] * c - ot[j] * sn)
                       : (xs[j] * c + ot[j] * sn);
    out[j] = (bf16)(o * outmul);
  }
  *(v4bf*)(Y + (((size_t)(b * H + h)) * S_ + s) * HD_ + lane * 4) = out;
}

// ---------------- causal flash attention, one wave per 16-row q tile ----------------
// Q bf16 [B,NH,S,HD]; K bf16 [B,NKV,S,HD]; Vt bf16 [B,NKV,HD,S];
// O bf16 [B,S,NH*HD].
__global__ void flash_attn_wmma(const bf16* __restrict__ Q,
                                const bf16* __restrict__ Kd,
                                const bf16* __restrict__ Vt,
                                bf16* __restrict__ O) {
  __shared__ __align__(16) bf16 Plds[16 * 32];
  const int lane = threadIdx.x;
  const int hi = lane >> 4, lm = lane & 15;
  const int qt = blockIdx.x;          // S/16 q tiles
  const int h  = blockIdx.y;          // NH
  const int b  = blockIdx.z;          // B
  const int hkv = h / (NH_ / NKV_);
  const int qbase = qt * 16;

  const bf16* qrow = Q + (((size_t)(b * NH_ + h)) * S_ + qbase + lm) * HD_;
  v16bf qf[4];
#pragma unroll
  for (int c = 0; c < 4; ++c) qf[c] = wmma_a_frag(qrow + c * 32, hi);

  v8f o[8] = {};
  float mstat[8], lstat[8];
#pragma unroll
  for (int r = 0; r < 8; ++r) { mstat[r] = -1e30f; lstat[r] = 0.0f; }

  const bf16* kbase = Kd + ((size_t)(b * NKV_ + hkv)) * S_ * HD_;
  const bf16* vbase = Vt + ((size_t)(b * NKV_ + hkv)) * HD_ * S_;

  for (int kc = 0; kc < qbase + 16; kc += 32) {
    if (kc + 32 < qbase + 16) {        // gfx1250 global_prefetch of next K tile
      __builtin_prefetch(kbase + (size_t)(kc + 32) * HD_, 0, 1);
    }
    // ---- scores: Q (16x128) x K^T (128x32) via 2 key tiles x 4 HD chunks ----
    v8f sc[2] = {};
#pragma unroll
    for (int t = 0; t < 2; ++t) {
      const bf16* kp = kbase + (size_t)(kc + t * 16 + lm) * HD_;
#pragma unroll
      for (int c = 0; c < 4; ++c) {
        v16bf kfrag = *(const v16bf*)(kp + c * 32 + hi * 16);
        sc[t] = wmma_bf16(qf[c], kfrag, sc[t]);
      }
    }
    const int key0 = kc + lm, key1 = kc + 16 + lm;

    // ---- causal mask + online softmax (row stats live per half-wave) ----
#pragma unroll
    for (int r = 0; r < 8; ++r) {
      int row = qbase + r + 8 * hi;
      float s0 = (key0 <= row) ? sc[0][r] : -1e30f;
      float s1 = (key1 <= row) ? sc[1][r] : -1e30f;
      float mx = fmaxf(s0, s1);
#pragma unroll
      for (int m = 8; m >= 1; m >>= 1) mx = fmaxf(mx, __shfl_xor(mx, m, 32));
      float mnew  = fmaxf(mstat[r], mx);
      float alpha = __expf(mstat[r] - mnew);
      float p0 = (key0 <= row) ? __expf(s0 - mnew) : 0.0f;
      float p1 = (key1 <= row) ? __expf(s1 - mnew) : 0.0f;
      float rsum = p0 + p1;
#pragma unroll
      for (int m = 8; m >= 1; m >>= 1) rsum += __shfl_xor(rsum, m, 32);
      lstat[r] = lstat[r] * alpha + rsum;
      mstat[r] = mnew;
#pragma unroll
      for (int n = 0; n < 8; ++n) o[n][r] *= alpha;
      Plds[(r + 8 * hi) * 32 + lm]      = (bf16)p0;
      Plds[(r + 8 * hi) * 32 + 16 + lm] = (bf16)p1;
    }
    __syncthreads();
    v16bf pf = wmma_a_frag(&Plds[lm * 32], hi);   // P as 16x32 A fragment
    __syncthreads();

    // ---- O += P (16x32) x V (32x128): 8 output-dim tiles ----
#pragma unroll
    for (int n = 0; n < 8; ++n) {
      const bf16* vp = vbase + (size_t)(n * 16 + lm) * S_ + kc + hi * 16;
      v16bf vfrag = *(const v16bf*)vp;
      o[n] = wmma_bf16(pf, vfrag, o[n]);
    }
  }

  // ---- epilogue: divide by l, store bf16 [B,S,NH*HD] ----
#pragma unroll
  for (int r = 0; r < 8; ++r) {
    float inv = 1.0f / lstat[r];
    int row = qbase + r + 8 * hi;
    bf16* op = O + ((size_t)(b * S_ + row)) * (NH_ * HD_) + h * HD_;
#pragma unroll
    for (int n = 0; n < 8; ++n) op[n * 16 + lm] = (bf16)(o[n][r] * inv);
  }
}

// ---------------- host-side orchestration ----------------
extern "C" void kernel_launch(void* const* d_in, const int* in_sizes, int n_in,
                              void* d_out, int out_size, void* d_ws, size_t ws_size,
                              hipStream_t stream) {
  const float* hidden    = (const float*)d_in[0];
  const int*   positions = (const int*)  d_in[1];
  const float* Wq        = (const float*)d_in[2];
  const float* Wk        = (const float*)d_in[3];
  const float* Wv        = (const float*)d_in[4];
  const float* Wc        = (const float*)d_in[5];
  const float* q_scale   = (const float*)d_in[6];
  const float* k_scale   = (const float*)d_in[7];
  float* out = (float*)d_out;

  const int M = B_ * S_;              // 4096 token rows
  char* ws = (char*)d_ws;
  auto carve = [&](size_t bytes) -> char* {
    char* p = ws; ws += (bytes + 255) & ~(size_t)255; return p;
  };
  bf16*  hid_bf = (bf16*)carve((size_t)M * HID_ * 2);
  bf16*  WqT    = (bf16*)carve((size_t)HID_ * NH_  * HD_ * 2);
  bf16*  WkT    = (bf16*)carve((size_t)HID_ * NKV_ * HD_ * 2);
  bf16*  WvT    = (bf16*)carve((size_t)HID_ * NKV_ * HD_ * 2);
  bf16*  WcT    = (bf16*)carve((size_t)(NH_ * HD_) * HID_ * 2);
  float* qf32   = (float*)carve((size_t)M * NH_  * HD_ * 4);
  float* kf32   = (float*)carve((size_t)M * NKV_ * HD_ * 4);
  float* vf32   = (float*)carve((size_t)M * NKV_ * HD_ * 4);
  bf16*  q_bf   = (bf16*)carve((size_t)M * NH_  * HD_ * 2);
  bf16*  k_bf   = (bf16*)carve((size_t)M * NKV_ * HD_ * 2);
  bf16*  vT_bf  = (bf16*)carve((size_t)M * NKV_ * HD_ * 2);
  bf16*  att_bf = (bf16*)carve((size_t)M * NH_  * HD_ * 2);

  // 1) convert hidden + weights to bf16 (weights transposed to [N,K])
  {
    int n = M * HID_;
    cvt_f32_to_bf16<<<(n + 255) / 256, 256, 0, stream>>>(hidden, hid_bf, n);
  }
  {
    int n = HID_ * NH_ * HD_;
    transpose_to_bf16<<<(n + 255) / 256, 256, 0, stream>>>(Wq, WqT, HID_, NH_ * HD_);
  }
  {
    int n = HID_ * NKV_ * HD_;
    transpose_to_bf16<<<(n + 255) / 256, 256, 0, stream>>>(Wk, WkT, HID_, NKV_ * HD_);
    transpose_to_bf16<<<(n + 255) / 256, 256, 0, stream>>>(Wv, WvT, HID_, NKV_ * HD_);
  }
  {
    int n = NH_ * HD_ * HID_;
    transpose_to_bf16<<<(n + 255) / 256, 256, 0, stream>>>(Wc, WcT, NH_ * HD_, HID_);
  }

  // 2) QKV projections (LDS-tiled WMMA with async global->LDS staging)
  wmma_gemm_tiled<<<dim3(M / TM, (NH_ * HD_) / TN), 256, 0, stream>>>(
      hid_bf, WqT, qf32, M, NH_ * HD_, HID_);
  wmma_gemm_tiled<<<dim3(M / TM, (NKV_ * HD_) / TN), 256, 0, stream>>>(
      hid_bf, WkT, kf32, M, NKV_ * HD_, HID_);
  wmma_gemm_tiled<<<dim3(M / TM, (NKV_ * HD_) / TN), 256, 0, stream>>>(
      hid_bf, WvT, vf32, M, NKV_ * HD_, HID_);

  // 3) RMSNorm + RoPE (q gets HD^-0.5 folded in); V transpose/convert
  {
    int nrows = M * NH_;   // one wave32 per (b,s,head)
    norm_rope_kernel<<<(nrows * 32 + 255) / 256, 256, 0, stream>>>(
        qf32, positions, q_scale, q_bf, NH_, nrows, QK_SCALE);
  }
  {
    int nrows = M * NKV_;
    norm_rope_kernel<<<(nrows * 32 + 255) / 256, 256, 0, stream>>>(
        kf32, positions, k_scale, k_bf, NKV_, nrows, 1.0f);
  }
  {
    int n = M * NKV_ * HD_;
    v_transpose_bf16<<<(n + 255) / 256, 256, 0, stream>>>(vf32, vT_bf);
  }

  // 4) causal flash attention (WMMA QK^T and PV)
  flash_attn_wmma<<<dim3(S_ / 16, NH_, B_), 32, 0, stream>>>(q_bf, k_bf, vT_bf, att_bf);

  // 5) output projection (LDS-tiled WMMA) -> fp32 d_out
  wmma_gemm_tiled<<<dim3(M / TM, HID_ / TN), 256, 0, stream>>>(
      att_bf, WcT, out, M, HID_, NH_ * HD_);
}
